// PhyFreqVSSD_20993800143241
// MI455X (gfx1250) — matrix operand order repaired and checked by reading, not compiled, not run on hardware
//
#include <hip/hip_runtime.h>

// ---------------------------------------------------------------------------
// Types for CDNA5 WMMA (wave32)
// ---------------------------------------------------------------------------
typedef __bf16 bf16_t;
typedef bf16_t v16bf __attribute__((ext_vector_type(16)));
typedef bf16_t bfx8  __attribute__((ext_vector_type(8)));
typedef float  v8f   __attribute__((ext_vector_type(8)));

#define DEV __device__ __forceinline__

DEV float sigmf(float v) { return 1.0f / (1.0f + __expf(-v)); }
DEV float geluf(float v) {
    // jax.nn.gelu (approximate=True): 0.5*x*(1+tanh(sqrt(2/pi)*(x+0.044715*x^3)))
    const float c = 0.7978845608028654f;
    float u = c * (v + 0.044715f * v * v * v);
    return 0.5f * v * (1.0f + tanhf(u));
}

// Row statistics across a 256-thread block (8 wave32s), 2 values per thread.
DEV void row_stats(float v0, float v1, float D, float& mu, float& rs) {
    float s  = v0 + v1;
    float ss = v0 * v0 + v1 * v1;
#pragma unroll
    for (int off = 16; off > 0; off >>= 1) {
        s  += __shfl_xor(s,  off, 32);
        ss += __shfl_xor(ss, off, 32);
    }
    __shared__ float red[16];
    int wid = threadIdx.x >> 5, ln = threadIdx.x & 31;
    if (ln == 0) { red[wid] = s; red[8 + wid] = ss; }
    __syncthreads();
    float ts = 0.f, tss = 0.f;
#pragma unroll
    for (int i = 0; i < 8; ++i) { ts += red[i]; tss += red[8 + i]; }
    mu = ts / D;
    float var = tss / D - mu * mu;
    rs = rsqrtf(var + 1e-5f);
    __syncthreads();
}

// ---------------------------------------------------------------------------
// Weight prep: fp32 W[R][C] -> bf16 Wt[C][R]  (Wt[col][k] layout for WMMA B)
// ---------------------------------------------------------------------------
__global__ void wprep_kernel(const float* __restrict__ W, bf16_t* __restrict__ Wt,
                             int R, int C) {
    int idx = blockIdx.x * blockDim.x + threadIdx.x;
    if (idx >= R * C) return;
    int r = idx / C, c = idx % C;
    Wt[(size_t)c * R + r] = (bf16_t)W[(size_t)r * C + c];
}

// ---------------------------------------------------------------------------
// LayerNorm: x -> h (f32) and h (bf16).  One block (256 thr) per row of 512.
// ---------------------------------------------------------------------------
__global__ __launch_bounds__(256)
void ln_kernel(const float* __restrict__ x, const float* __restrict__ w,
               const float* __restrict__ b, float* __restrict__ hf,
               bf16_t* __restrict__ hb) {
    const int Dm = 512;
    size_t base = (size_t)blockIdx.x * Dm;
    int t = threadIdx.x;
    float v0 = x[base + t], v1 = x[base + t + 256];
    float mu, rs;
    row_stats(v0, v1, (float)Dm, mu, rs);
    float o0 = (v0 - mu) * rs * w[t] + b[t];
    float o1 = (v1 - mu) * rs * w[t + 256] + b[t + 256];
    hf[base + t] = o0;          hf[base + t + 256] = o1;
    hb[base + t] = (bf16_t)o0;  hb[base + t + 256] = (bf16_t)o1;
}

// ---------------------------------------------------------------------------
// Selective scan: serial over L, parallel over B*D channels.
// delta/b/c in bf16; writes ssd (f32) and bf16 into fused[:, 0:512] (ld 1024).
// ---------------------------------------------------------------------------
__global__ __launch_bounds__(256)
void scan_kernel(const float* __restrict__ h, const bf16_t* __restrict__ dlt,
                 const bf16_t* __restrict__ bt, const bf16_t* __restrict__ ct,
                 const float* __restrict__ skip, float* __restrict__ ssd,
                 bf16_t* __restrict__ fused, int L) {
    const int Dm = 512;
    int ch = blockIdx.x * blockDim.x + threadIdx.x;   // 0 .. B*D-1
    int bb = ch / Dm, d = ch % Dm;
    float sk = skip[d];
    float state = 0.0f;
    size_t base = (size_t)bb * L * Dm + d;
    size_t rowb = (size_t)bb * L;
    for (int t = 0; t < L; ++t) {
        size_t i = base + (size_t)t * Dm;
        float xt  = h[i];
        float dtv = (float)dlt[i];
        float btv = (float)bt[i];
        float ctv = (float)ct[i];
        state = sigmf(dtv) * state + tanhf(btv) * xt;
        float out = tanhf(ctv) * state + sk * xt;
        ssd[i] = out;
        fused[(rowb + t) * 1024 + d] = (bf16_t)out;
    }
}

// ---------------------------------------------------------------------------
// mix + residual + LayerNorm:
//   y = x + g*ssd + (1-g)*mamba ; write y (f32, d_out) and LN(y) in bf16
// ---------------------------------------------------------------------------
__global__ __launch_bounds__(256)
void mixln_kernel(const float* __restrict__ x, const float* __restrict__ gate,
                  const float* __restrict__ ssd, const float* __restrict__ mam,
                  const float* __restrict__ w, const float* __restrict__ b,
                  float* __restrict__ y, bf16_t* __restrict__ ylnb) {
    const int Dm = 512;
    size_t base = (size_t)blockIdx.x * Dm;
    int t = threadIdx.x;
    float g0 = gate[base + t],       g1 = gate[base + t + 256];
    float y0 = x[base + t]       + g0 * ssd[base + t]       + (1.f - g0) * mam[base + t];
    float y1 = x[base + t + 256] + g1 * ssd[base + t + 256] + (1.f - g1) * mam[base + t + 256];
    float mu, rs;
    row_stats(y0, y1, (float)Dm, mu, rs);
    y[base + t] = y0;  y[base + t + 256] = y1;
    ylnb[base + t]       = (bf16_t)((y0 - mu) * rs * w[t]       + b[t]);
    ylnb[base + t + 256] = (bf16_t)((y1 - mu) * rs * w[t + 256] + b[t + 256]);
}

// ---------------------------------------------------------------------------
// WMMA GEMM:  out = epilogue( A[rows x K] @ W[K x cols] + bias )
//   A   : bf16 row-major
//   Wt  : bf16, W transposed -> Wt[col][k], row-major [cols x K]
// Block: 256 threads = 8 waves arranged 2(M) x 4(N); block tile 64 x 256.
// Wave tile 32 x 64: 2 A-fragments x 4 B-fragments = 8 accumulators; every
// B-fragment load feeds 2 WMMAs (12 b128 loads per 8 WMMAs per K-step).
// Epilogue modes:
//   1: outB = bf16(v)
//   2: outB = bf16(gelu(v))
//   3: router: g=sigmoid(v); outB=bf16(aux*g); outB2=bf16(aux*(1-g))
//   4: outF = sigmoid(v)
//   5: outF = v ; outB = bf16(v)            (mamba out -> fused[:,512:])
//   6: outF = aux + v                        (final: y + ffn)
// ---------------------------------------------------------------------------
__global__ __launch_bounds__(256)
void gemm_bf16_wmma(const bf16_t* __restrict__ A, const bf16_t* __restrict__ Wt,
                    const float* __restrict__ bias, float* __restrict__ outF,
                    bf16_t* __restrict__ outB, const float* __restrict__ aux,
                    bf16_t* __restrict__ outB2, int K, int cols, int ldB,
                    int colOff, int mode) {
    const int tid    = threadIdx.x;
    const int wid    = tid >> 5;
    const int lane   = tid & 31;
    const int laneM  = lane & 15;
    const int laneHi = lane >> 4;
    const int rowBase = blockIdx.y * 64 + (wid & 1) * 32;
    const int colBase = blockIdx.x * 256 + (wid >> 1) * 64;

    v8f acc[2][4] = {};

    // A fragment addressing (16-bit A 16x32 layout):
    //   lanes 0-15 : K in {0..7, 16..23};  lanes 16-31 : K in {8..15, 24..31}
    const bf16_t* aPtr0 = A + (size_t)(rowBase + laneM) * K + (size_t)laneHi * 8;
    const bf16_t* aPtr1 = aPtr0 + (size_t)16 * K;

    // B fragment addressing (16-bit B 32x16 layout):
    //   lanes 0-15 : K in {0..7, 16..23};  lanes 16-31 : K in {8..15, 24..31}
    const bf16_t* bPtr[4];
#pragma unroll
    for (int j = 0; j < 4; ++j)
        bPtr[j] = Wt + (size_t)(colBase + j * 16 + laneM) * K + (size_t)laneHi * 8;

    for (int k0 = 0; k0 < K; k0 += 32) {
        bfx8 a00 = *(const bfx8*)(aPtr0 + k0);
        bfx8 a01 = *(const bfx8*)(aPtr0 + k0 + 16);
        bfx8 a10 = *(const bfx8*)(aPtr1 + k0);
        bfx8 a11 = *(const bfx8*)(aPtr1 + k0 + 16);
        v16bf af0, af1;
#pragma unroll
        for (int i = 0; i < 8; ++i) {
            af0[i] = a00[i]; af0[i + 8] = a01[i];
            af1[i] = a10[i]; af1[i + 8] = a11[i];
        }
#pragma unroll
        for (int j = 0; j < 4; ++j) {
            bfx8 b0 = *(const bfx8*)(bPtr[j] + k0);
            bfx8 b1 = *(const bfx8*)(bPtr[j] + k0 + 16);
            v16bf bfrag;
#pragma unroll
            for (int i = 0; i < 8; ++i) { bfrag[i] = b0[i]; bfrag[i + 8] = b1[i]; }
            acc[0][j] = __builtin_amdgcn_wmma_f32_16x16x32_bf16(
                false, af0, false, bfrag, (short)0, acc[0][j], false, false);
            acc[1][j] = __builtin_amdgcn_wmma_f32_16x16x32_bf16(
                false, af1, false, bfrag, (short)0, acc[1][j], false, false);
        }
    }

    // Epilogue. C/D layout: lanes 0-15 N=lane, M=r; lanes 16-31 N=lane-16, M=8+r.
#pragma unroll
    for (int mi = 0; mi < 2; ++mi) {
#pragma unroll
        for (int j = 0; j < 4; ++j) {
            int col = colBase + j * 16 + laneM;
            float bsv = bias[col];
#pragma unroll
            for (int r = 0; r < 8; ++r) {
                int row = rowBase + mi * 16 + r + laneHi * 8;
                float v = acc[mi][j][r] + bsv;
                size_t oi = (size_t)row * cols + col;
                size_t ob = (size_t)row * ldB + colOff + col;
                switch (mode) {
                    case 1: outB[ob] = (bf16_t)v; break;
                    case 2: outB[ob] = (bf16_t)geluf(v); break;
                    case 3: {
                        float g = sigmf(v), hv = aux[oi];
                        outB[ob]  = (bf16_t)(hv * g);
                        outB2[ob] = (bf16_t)(hv * (1.0f - g));
                    } break;
                    case 4: outF[oi] = sigmf(v); break;
                    case 5: outF[oi] = v; outB[ob] = (bf16_t)v; break;
                    case 6: outF[oi] = aux[oi] + v; break;
                    default: outF[oi] = v; break;
                }
            }
        }
    }
}

// ---------------------------------------------------------------------------
// Host orchestration
// ---------------------------------------------------------------------------
extern "C" void kernel_launch(void* const* d_in, const int* in_sizes, int n_in,
                              void* d_out, int out_size, void* d_ws, size_t ws_size,
                              hipStream_t stream) {
    const int Bb = 8, L = 2048, Dm = 512;
    const int N = Bb * L;              // 16384 rows
    const int H1 = 1024, H2 = 2048;

    const float* x    = (const float*)d_in[0];
    const float* ln_w = (const float*)d_in[1];
    const float* ln_b = (const float*)d_in[2];
    const float* rw   = (const float*)d_in[3];
    const float* rb   = (const float*)d_in[4];
    const float* dw   = (const float*)d_in[5];
    const float* db   = (const float*)d_in[6];
    const float* bw   = (const float*)d_in[7];
    const float* bb   = (const float*)d_in[8];
    const float* cw   = (const float*)d_in[9];
    const float* cb   = (const float*)d_in[10];
    const float* skip = (const float*)d_in[11];
    const float* mw1  = (const float*)d_in[12];
    const float* mb1  = (const float*)d_in[13];
    const float* mw2  = (const float*)d_in[14];
    const float* mb2  = (const float*)d_in[15];
    const float* gw   = (const float*)d_in[16];
    const float* gb   = (const float*)d_in[17];
    const float* fln_w= (const float*)d_in[18];
    const float* fln_b= (const float*)d_in[19];
    const float* fw1  = (const float*)d_in[20];
    const float* fb1  = (const float*)d_in[21];
    const float* fw2  = (const float*)d_in[22];
    const float* fb2  = (const float*)d_in[23];
    float* out = (float*)d_out;

    // ---- workspace layout ----
    char* ws = (char*)d_ws;
    size_t off = 0;
    auto alloc = [&](size_t bytes) -> char* {
        char* p = ws + off;
        off += (bytes + 255) & ~(size_t)255;
        return p;
    };
    const size_t ND  = (size_t)N * Dm;
    const size_t NH1 = (size_t)N * H1;
    const size_t NH2 = (size_t)N * H2;

    bf16_t* rwT  = (bf16_t*)alloc((size_t)Dm * Dm * 2);
    bf16_t* dwT  = (bf16_t*)alloc((size_t)Dm * Dm * 2);
    bf16_t* bwT  = (bf16_t*)alloc((size_t)Dm * Dm * 2);
    bf16_t* cwT  = (bf16_t*)alloc((size_t)Dm * Dm * 2);
    bf16_t* mw1T = (bf16_t*)alloc((size_t)Dm * H1 * 2);
    bf16_t* mw2T = (bf16_t*)alloc((size_t)H1 * Dm * 2);
    bf16_t* gwT  = (bf16_t*)alloc((size_t)H1 * Dm * 2);
    bf16_t* fw1T = (bf16_t*)alloc((size_t)Dm * H2 * 2);
    bf16_t* fw2T = (bf16_t*)alloc((size_t)H2 * Dm * 2);

    float*  h_f32   = (float*) alloc(ND * 4);
    bf16_t* h_bf    = (bf16_t*)alloc(ND * 2);
    bf16_t* low_bf  = (bf16_t*)alloc(ND * 2);
    bf16_t* high_bf = (bf16_t*)alloc(ND * 2);
    bf16_t* delt_bf = (bf16_t*)alloc(ND * 2);
    bf16_t* bt_bf   = (bf16_t*)alloc(ND * 2);
    bf16_t* ct_bf   = (bf16_t*)alloc(ND * 2);
    float*  ssd_f32 = (float*) alloc(ND * 4);
    bf16_t* fused   = (bf16_t*)alloc(NH1 * 2);   // [N x 1024]: [ssd | mamba]
    bf16_t* mhid_bf = (bf16_t*)alloc(NH1 * 2);
    float*  mam_f32 = (float*) alloc(ND * 4);
    float*  gat_f32 = (float*) alloc(ND * 4);
    bf16_t* yln_bf  = (bf16_t*)alloc(ND * 2);
    bf16_t* ffnh_bf = (bf16_t*)alloc(NH2 * 2);
    (void)ws_size; (void)in_sizes; (void)n_in; (void)out_size;

    auto wprep = [&](const float* W, bf16_t* Wt, int R, int C) {
        int tot = R * C;
        wprep_kernel<<<(tot + 255) / 256, 256, 0, stream>>>(W, Wt, R, C);
    };
    wprep(rw,  rwT,  Dm, Dm);
    wprep(dw,  dwT,  Dm, Dm);
    wprep(bw,  bwT,  Dm, Dm);
    wprep(cw,  cwT,  Dm, Dm);
    wprep(mw1, mw1T, Dm, H1);
    wprep(mw2, mw2T, H1, Dm);
    wprep(gw,  gwT,  H1, Dm);
    wprep(fw1, fw1T, Dm, H2);
    wprep(fw2, fw2T, H2, Dm);

    auto gemm = [&](const bf16_t* A, const bf16_t* Wt, const float* bias,
                    float* outF, bf16_t* outB, const float* aux, bf16_t* outB2,
                    int K, int cols, int ldB, int colOff, int mode) {
        dim3 grid(cols / 256, N / 64);
        gemm_bf16_wmma<<<grid, 256, 0, stream>>>(A, Wt, bias, outF, outB, aux,
                                                 outB2, K, cols, ldB, colOff, mode);
    };

    // 1. h = LN(x)
    ln_kernel<<<N, 256, 0, stream>>>(x, ln_w, ln_b, h_f32, h_bf);
    // 2. router: g = sigmoid(h@rw+rb); low = h*g; high = h*(1-g)
    gemm(h_bf, rwT, rb, nullptr, low_bf, h_f32, high_bf, Dm, Dm, Dm, 0, 3);
    // 3. delta / b_term / c_term (bf16)
    gemm(low_bf,  dwT, db, nullptr, delt_bf, nullptr, nullptr, Dm, Dm, Dm, 0, 1);
    gemm(low_bf,  bwT, bb, nullptr, bt_bf,   nullptr, nullptr, Dm, Dm, Dm, 0, 1);
    gemm(high_bf, cwT, cb, nullptr, ct_bf,   nullptr, nullptr, Dm, Dm, Dm, 0, 1);
    // 4. mamba hidden = gelu(low@mw1+mb1)
    gemm(low_bf, mw1T, mb1, nullptr, mhid_bf, nullptr, nullptr, Dm, H1, H1, 0, 2);
    // 5. selective scan -> ssd_f32, fused[:, :512]
    scan_kernel<<<(Bb * Dm) / 256, 256, 0, stream>>>(h_f32, delt_bf, bt_bf, ct_bf,
                                                     skip, ssd_f32, fused, L);
    // 6. mamba out = hidden@mw2+mb2 -> f32 + fused[:, 512:]
    gemm(mhid_bf, mw2T, mb2, mam_f32, fused, nullptr, nullptr, H1, Dm, H1, Dm, 5);
    // 7. gate = sigmoid(fused@gw+gb)
    gemm(fused, gwT, gb, gat_f32, nullptr, nullptr, nullptr, H1, Dm, Dm, 0, 4);
    // 8. y = x + g*ssd + (1-g)*mamba ; yln = LN(y)
    mixln_kernel<<<N, 256, 0, stream>>>(x, gat_f32, ssd_f32, mam_f32, fln_w, fln_b,
                                        out, yln_bf);
    // 9. ffn hidden = gelu(yln@fw1+fb1)
    gemm(yln_bf, fw1T, fb1, nullptr, ffnh_bf, nullptr, nullptr, Dm, H2, H2, 0, 2);
    // 10. out = y + ffnh@fw2 + fb2
    gemm(ffnh_bf, fw2T, fb2, out, nullptr, out, nullptr, H2, Dm, Dm, 0, 6);
}
